// EdgeConvNet_63513976373543
// MI455X (gfx1250) — matrix operand order repaired and checked by reading, not compiled, not run on hardware
//
#include <hip/hip_runtime.h>
#include <hip/hip_bf16.h>

typedef __attribute__((ext_vector_type(16))) _Float16 v16h;
typedef __attribute__((ext_vector_type(8)))  _Float16 v8h;
typedef __attribute__((ext_vector_type(8)))  float    v8f;
typedef __attribute__((ext_vector_type(4)))  float    v4f;

#define EPSBN 1e-5f

// load 8 consecutive f32 (32B, 16B-aligned) as two b128 loads
__device__ __forceinline__ void load8f(const float* __restrict__ p, float* o) {
  const v4f a = *(const v4f*)p;
  const v4f b = *(const v4f*)(p + 4);
#pragma unroll
  for (int i = 0; i < 4; ++i) { o[i] = a[i]; o[4 + i] = b[i]; }
}

// h after conv1 = [A1(64) | x(16)];  after conv2 = [A2(64) | A1(64) | x(16)]
// All region boundaries are 8-aligned; k0 is always an 8-aligned run start and
// a run never straddles a boundary, so one base pointer covers the whole run.
__device__ __forceinline__ const float* p80(const float* __restrict__ X,
                                            const float* __restrict__ A1,
                                            int n, int k0) {
  return (k0 < 64) ? (A1 + n * 64 + k0) : (X + n * 16 + (k0 - 64));
}

__device__ __forceinline__ const float* p144(const float* __restrict__ X,
                                             const float* __restrict__ A1,
                                             const float* __restrict__ A2,
                                             const float* __restrict__ zbuf,
                                             int n, int k0) {
  if (k0 < 64)  return A2 + n * 64 + k0;
  if (k0 < 128) return A1 + n * 64 + (k0 - 64);
  if (k0 < 144) return X + n * 16 + (k0 - 128);
  return zbuf;                       // K padded to a multiple of 32
}

// ---------------------------------------------------------------------------
// Fused gather + Linear (WMMA f16, f32 accum) + bias -> raw pre-BN f16 out,
// accumulating per-column sum / sum-of-squares for the following batchnorm.
// One wave = 16 rows x NC cols; K-chunks of 32, fully unrolled (K constexpr).
// MODE 0: conv1 layer0 gather [xi | xj-xi], in=16  (K=32)
// MODE 1: conv2 layer0 gather [xi | xj-xi], in=80  (K=160)
// MODE 2: read prev f16 buffer, folded BN+ReLU     (K=64)
// MODE 3: node head input (144-dim node features)  (K=144)
// MODE 4: edge head input h[src]-h[dst]            (K=144)
// ---------------------------------------------------------------------------
template <int MODE, int K, int NCT>
__global__ __launch_bounds__(256) void mlp_wmma_kernel(
    const float* __restrict__ W,      // [K, NC] row-major
    const float* __restrict__ bias,   // [NC]
    const float* __restrict__ scale,  // folded-BN scale (MODE 2)
    const float* __restrict__ shift,  // folded-BN shift (MODE 2)
    const _Float16* __restrict__ prev,
    const float* __restrict__ X, const float* __restrict__ A1f,
    const float* __restrict__ A2f, const float* __restrict__ zbuf,
    const long long* __restrict__ srcIdx, const long long* __restrict__ dstIdx,
    _Float16* __restrict__ outT,      // [rows, NC] raw pre-BN (f16)
    float* __restrict__ ssum, float* __restrict__ ssqs,
    int rows) {
  constexpr int NC = NCT * 16;
  constexpr int CH = (K + 31) / 32;          // K-chunks of 32
  __shared__ __align__(32) _Float16 sW[CH * NCT * 512];  // fragment-order weights
  __shared__ __align__(16) _Float16 stile[8][256];        // per-wave 16x16 D tile

  const int tid = threadIdx.x;
  // Pre-swizzle W into WMMA B-fragment order:
  //   frag (chunk c, tile t): lane l holds col = t*16+(l&15),
  //   K = c*32 + (e>>3)*16 + (l>>4)*8 + (e&7)  for e = 0..15  (table order)
#pragma unroll 1
  for (int i = tid; i < CH * NCT * 512; i += 256) {
    const int c   = i / (NCT * 512);
    const int rem = i - c * (NCT * 512);
    const int t   = rem >> 9;
    const int r2  = rem & 511;
    const int l   = r2 >> 4;
    const int e   = r2 & 15;
    const int k   = c * 32 + (e >> 3) * 16 + (l >> 4) * 8 + (e & 7);
    const int col = t * 16 + (l & 15);
    sW[i] = (k < K) ? (_Float16)W[k * NC + col] : (_Float16)0.f;
  }
  __syncthreads();

  const int lane  = tid & 31;
  const int wave  = tid >> 5;
  const int tile0 = blockIdx.x * 128 + wave * 16;
  const int rowL  = lane & 15;   // A-frag: this lane gathers row rowL
  const int grp   = lane >> 4;   // K-half selector (ISA 16-bit A layout)
  int myRow = tile0 + rowL;
  if (myRow >= rows) myRow = rows - 1;   // clamp: invalid rows discarded later

  int si = 0, di = 0;
  if constexpr (MODE == 0 || MODE == 1 || MODE == 4) {
    si = (int)srcIdx[myRow];
    di = (int)dstIdx[myRow];
  }

  v8f acc[NCT];
#pragma unroll
  for (int t = 0; t < NCT; ++t)
#pragma unroll
    for (int i = 0; i < 8; ++i) acc[t][i] = 0.f;

#pragma unroll
  for (int c = 0; c < CH; ++c) {
    // ---- A fragment: two contiguous 8-wide runs per lane ----
    v16h a;
#pragma unroll
    for (int half = 0; half < 2; ++half) {
      const int kbase = c * 32 + half * 16 + grp * 8;   // run start (8-aligned)
      float r8[8];
      if constexpr (MODE == 0) {
        if (half == 0) {                                 // xi region (k<16)
          load8f(X + di * 16 + kbase, r8);
        } else {                                         // xj - xi
          float t1[8], t2[8];
          load8f(X + si * 16 + (kbase - 16), t1);
          load8f(X + di * 16 + (kbase - 16), t2);
#pragma unroll
          for (int i = 0; i < 8; ++i) r8[i] = t1[i] - t2[i];
        }
      } else if constexpr (MODE == 1) {
        const bool xi = (kbase < 80);                    // constant-folds per run
        const int  kk = xi ? kbase : (kbase - 80);
        const float* pa = xi ? p80(X, A1f, di, kk) : p80(X, A1f, si, kk);
        const float* pb = xi ? zbuf : p80(X, A1f, di, kk);
        float t1[8], t2[8];
        load8f(pa, t1);
        load8f(pb, t2);
#pragma unroll
        for (int i = 0; i < 8; ++i) r8[i] = t1[i] - t2[i];
      } else if constexpr (MODE == 2) {
        const v8h hv = *(const v8h*)(prev + myRow * 64 + kbase);
#pragma unroll
        for (int i = 0; i < 8; ++i)
          r8[i] = fmaxf(scale[kbase + i] * (float)hv[i] + shift[kbase + i], 0.f);
      } else if constexpr (MODE == 3) {
        load8f(p144(X, A1f, A2f, zbuf, myRow, kbase), r8);
      } else {  // MODE 4
        float t1[8], t2[8];
        load8f(p144(X, A1f, A2f, zbuf, si, kbase), t1);
        load8f(p144(X, A1f, A2f, zbuf, di, kbase), t2);
#pragma unroll
        for (int i = 0; i < 8; ++i) r8[i] = t1[i] - t2[i];
      }
#pragma unroll
      for (int i = 0; i < 8; ++i) a[half * 8 + i] = (_Float16)r8[i];
    }
    // ---- B fragments: contiguous 32B per lane from pre-swizzled LDS ----
#pragma unroll
    for (int t = 0; t < NCT; ++t) {
      const v16h bf = *(const v16h*)(sW + (c * NCT + t) * 512 + lane * 16);
      acc[t] = __builtin_amdgcn_wmma_f32_16x16x32_f16(
          false, a, false, bf, (short)0, acc[t], false, false);
    }
  }

  // ---- epilogue: +bias, stage 16x16 tile in LDS (col-major), coalesced
  //      b128 global store, column stats (sum, sum^2) ----
  _Float16* myT = stile[wave];
  const int rdRow = lane >> 1;          // readback: row within tile
  const int rdSeg = lane & 1;           // readback: 8-column segment
  const int grow  = tile0 + rdRow;      // global row for readback store

#pragma unroll
  for (int t = 0; t < NCT; ++t) {
    const float bi = bias[t * 16 + rowL];
    float csum = 0.f, csq = 0.f;
    v8h ph;
#pragma unroll
    for (int v = 0; v < 8; ++v) {
      const float h = acc[t][v] + bi;
      ph[v] = (_Float16)h;
      const float hm = ((tile0 + v + 8 * grp) < rows) ? h : 0.f;
      csum += hm;
      csq  += hm * hm;
    }
    // col-major stage: lane writes its 8 rows of column rowL in one b128
    *(v8h*)(myT + rowL * 16 + 8 * grp) = ph;

    csum += __shfl_xor(csum, 16, 32);   // fold the two row-groups per column
    csq  += __shfl_xor(csq, 16, 32);
    if (grp == 0) {
      atomicAdd(&ssum[t * 16 + rowL], csum);
      atomicAdd(&ssqs[t * 16 + rowL], csq);
    }

    // LDS is in-order within a wave (ISA 15.15); wait for the b128 stores,
    // then gather one row segment per lane and store 16B coalesced.
    // (WAR vs the next tile's stores is covered by in-order LDS issue.)
    asm volatile("s_wait_dscnt 0x0" ::: "memory");
    v8h rowv;
#pragma unroll
    for (int i = 0; i < 8; ++i) rowv[i] = myT[(rdSeg * 8 + i) * 16 + rdRow];
    if (grow < rows)
      *(v8h*)(outT + grow * NC + t * 16 + rdSeg * 8) = rowv;
  }
}

// ---------------------------------------------------------------------------
// helpers
// ---------------------------------------------------------------------------
__global__ void count_kernel(const long long* __restrict__ dst,
                             float* __restrict__ cnt, int E) {
  const int e = blockIdx.x * blockDim.x + threadIdx.x;
  if (e < E) atomicAdd(&cnt[(int)dst[e]], 1.f);
}

__global__ void bn_finalize_kernel(float* __restrict__ ssum, float* __restrict__ ssqs,
                                   const float* __restrict__ gamma,
                                   const float* __restrict__ beta,
                                   float* __restrict__ scale, float* __restrict__ shift,
                                   int C, float invR) {
  const int j = blockIdx.x * blockDim.x + threadIdx.x;
  if (j < C) {
    const float m   = ssum[j] * invR;
    const float var = fmaxf(ssqs[j] * invR - m * m, 0.f);
    const float sc  = gamma[j] * rsqrtf(var + EPSBN);
    scale[j] = sc;
    shift[j] = beta[j] - sc * m;
    ssum[j] = 0.f;   // ready for next layer
    ssqs[j] = 0.f;
  }
}

// scatter-mean numerator: relu(BN(msg)) atomically accumulated per dst node
__global__ void agg_kernel(const _Float16* __restrict__ t,
                           const float* __restrict__ scale,
                           const float* __restrict__ shift,
                           const long long* __restrict__ dst,
                           float* __restrict__ agg, int E) {
  const int idx = blockIdx.x * blockDim.x + threadIdx.x;
  if (idx < E * 64) {
    const int e = idx >> 6, j = idx & 63;
    const float v = fmaxf(scale[j] * (float)t[idx] + shift[j], 0.f);
    atomicAdd(&agg[(int)dst[e] * 64 + j], v);
  }
}

__global__ void div_kernel(float* __restrict__ agg, const float* __restrict__ cnt, int N) {
  const int idx = blockIdx.x * blockDim.x + threadIdx.x;
  if (idx < N * 64) agg[idx] /= fmaxf(cnt[idx >> 6], 1.f);
}

// 128 -> 1 + sigmoid, one wave per row (lane-parallel dot, shfl reduction)
__global__ void head2_kernel(const _Float16* __restrict__ t,
                             const float* __restrict__ scale,
                             const float* __restrict__ shift,
                             const float* __restrict__ W2,
                             const float* __restrict__ b2,
                             float* __restrict__ out, int rows) {
  const int gt = blockIdx.x * blockDim.x + threadIdx.x;
  const int r = gt >> 5, lane = gt & 31;
  if (r >= rows) return;
  float acc = 0.f;
#pragma unroll
  for (int j = lane; j < 128; j += 32) {
    const float h = fmaxf(scale[j] * (float)t[r * 128 + j] + shift[j], 0.f);
    acc += h * W2[j];
  }
#pragma unroll
  for (int o = 16; o > 0; o >>= 1) acc += __shfl_xor(acc, o, 32);
  if (lane == 0) out[r] = 1.f / (1.f + __expf(-(acc + b2[0])));
}

// ---------------------------------------------------------------------------
// launch
// d_in flatten assumption (jax pytree, sorted dict keys inside params):
//   0: x [N,16]
//   1..24:  convs[c][l] = (W, b, gamma, beta),  idx = 1 + (c*3+l)*4 + t
//   25..30: edge_head {W1, W2, b1, b2, beta, gamma}
//   31..36: node_head {W1, W2, b1, b2, beta, gamma}
//   37: edge_index [2,E] int64
// ---------------------------------------------------------------------------
extern "C" void kernel_launch(void* const* d_in, const int* in_sizes, int n_in,
                              void* d_out, int out_size, void* d_ws, size_t ws_size,
                              hipStream_t stream) {
  const float* X = (const float*)d_in[0];
  const long long* EI = (const long long*)d_in[n_in - 1];
  const int N = in_sizes[0] / 16;
  const int E = in_sizes[n_in - 1] / 2;
  const long long* srcI = EI;
  const long long* dstI = EI + E;

  auto P = [&](int i) { return (const float*)d_in[i]; };

  // ---- workspace layout ----
  char* base = (char*)d_ws;
  size_t off = 0;
  auto alloc = [&](size_t bytes) -> void* {
    void* p = base + off;
    off += (bytes + 255) & ~size_t(255);
    return p;
  };
  float* cnt  = (float*)alloc(sizeof(float) * N);
  float* A1   = (float*)alloc(sizeof(float) * (size_t)N * 64);
  float* A2   = (float*)alloc(sizeof(float) * (size_t)N * 64);
  float* ssum = (float*)alloc(sizeof(float) * 128);
  float* ssqs = (float*)alloc(sizeof(float) * 128);
  float* scal = (float*)alloc(sizeof(float) * 128);
  float* shft = (float*)alloc(sizeof(float) * 128);
  float* zbuf = (float*)alloc(sizeof(float) * 16);
  _Float16* tmpA = (_Float16*)alloc(sizeof(_Float16) * (size_t)E * 64);
  _Float16* tmpB = (_Float16*)alloc(sizeof(_Float16) * (size_t)E * 64);
  _Float16* tmpH = (_Float16*)alloc(sizeof(_Float16) * (size_t)E * 128);

  hipMemsetAsync(cnt, 0, sizeof(float) * N, stream);
  hipMemsetAsync(A1, 0, sizeof(float) * (size_t)N * 64, stream);
  hipMemsetAsync(A2, 0, sizeof(float) * (size_t)N * 64, stream);
  hipMemsetAsync(ssum, 0, sizeof(float) * 128, stream);
  hipMemsetAsync(ssqs, 0, sizeof(float) * 128, stream);
  hipMemsetAsync(zbuf, 0, sizeof(float) * 16, stream);

  count_kernel<<<(E + 255) / 256, 256, 0, stream>>>(dstI, cnt, E);

  const int gE = (E + 127) / 128;   // 128 rows per block (8 waves x 16)
  const int gN = (N + 127) / 128;
  const float invE = 1.f / (float)E;
  const float invN = 1.f / (float)N;

  // ================= conv 1 (in=16) =================
  mlp_wmma_kernel<0, 32, 4><<<gE, 256, 0, stream>>>(
      P(1), P(2), nullptr, nullptr, nullptr, X, nullptr, nullptr, zbuf,
      srcI, dstI, tmpA, ssum, ssqs, E);
  bn_finalize_kernel<<<1, 64, 0, stream>>>(ssum, ssqs, P(3), P(4), scal, shft, 64, invE);

  mlp_wmma_kernel<2, 64, 4><<<gE, 256, 0, stream>>>(
      P(5), P(6), scal, shft, tmpA, nullptr, nullptr, nullptr, zbuf,
      nullptr, nullptr, tmpB, ssum, ssqs, E);
  bn_finalize_kernel<<<1, 64, 0, stream>>>(ssum, ssqs, P(7), P(8), scal, shft, 64, invE);

  mlp_wmma_kernel<2, 64, 4><<<gE, 256, 0, stream>>>(
      P(9), P(10), scal, shft, tmpB, nullptr, nullptr, nullptr, zbuf,
      nullptr, nullptr, tmpA, ssum, ssqs, E);
  bn_finalize_kernel<<<1, 64, 0, stream>>>(ssum, ssqs, P(11), P(12), scal, shft, 64, invE);

  agg_kernel<<<(E * 64 + 255) / 256, 256, 0, stream>>>(tmpA, scal, shft, dstI, A1, E);
  div_kernel<<<(N * 64 + 255) / 256, 256, 0, stream>>>(A1, cnt, N);

  // ================= conv 2 (in=80) =================
  mlp_wmma_kernel<1, 160, 4><<<gE, 256, 0, stream>>>(
      P(13), P(14), nullptr, nullptr, nullptr, X, A1, nullptr, zbuf,
      srcI, dstI, tmpB, ssum, ssqs, E);
  bn_finalize_kernel<<<1, 64, 0, stream>>>(ssum, ssqs, P(15), P(16), scal, shft, 64, invE);

  mlp_wmma_kernel<2, 64, 4><<<gE, 256, 0, stream>>>(
      P(17), P(18), scal, shft, tmpB, nullptr, nullptr, nullptr, zbuf,
      nullptr, nullptr, tmpA, ssum, ssqs, E);
  bn_finalize_kernel<<<1, 64, 0, stream>>>(ssum, ssqs, P(19), P(20), scal, shft, 64, invE);

  mlp_wmma_kernel<2, 64, 4><<<gE, 256, 0, stream>>>(
      P(21), P(22), scal, shft, tmpA, nullptr, nullptr, nullptr, zbuf,
      nullptr, nullptr, tmpB, ssum, ssqs, E);
  bn_finalize_kernel<<<1, 64, 0, stream>>>(ssum, ssqs, P(23), P(24), scal, shft, 64, invE);

  agg_kernel<<<(E * 64 + 255) / 256, 256, 0, stream>>>(tmpB, scal, shft, dstI, A2, E);
  div_kernel<<<(N * 64 + 255) / 256, 256, 0, stream>>>(A2, cnt, N);

  // ================= node head (rows = N, in=144) =================
  mlp_wmma_kernel<3, 144, 8><<<gN, 256, 0, stream>>>(
      P(31), P(33), nullptr, nullptr, nullptr, X, A1, A2, zbuf,
      nullptr, nullptr, tmpH, ssum, ssqs, N);
  bn_finalize_kernel<<<1, 128, 0, stream>>>(ssum, ssqs, P(36), P(35), scal, shft, 128, invN);
  head2_kernel<<<((size_t)N * 32 + 255) / 256, 256, 0, stream>>>(
      tmpH, scal, shft, P(32), P(34), (float*)d_out, N);

  // ================= edge head (rows = E, in=144) =================
  mlp_wmma_kernel<4, 144, 8><<<gE, 256, 0, stream>>>(
      P(25), P(27), nullptr, nullptr, nullptr, X, A1, A2, zbuf,
      srcI, dstI, tmpH, ssum, ssqs, E);
  bn_finalize_kernel<<<1, 128, 0, stream>>>(ssum, ssqs, P(30), P(29), scal, shft, 128, invE);
  head2_kernel<<<((size_t)E * 32 + 255) / 256, 256, 0, stream>>>(
      tmpH, scal, shft, P(26), P(28), (float*)d_out + N, E);
}
